// F_ALSHConv2d_40896678592535
// MI455X (gfx1250) — matrix operand order repaired and checked by reading, not compiled, not run on hardware
//
#include <hip/hip_runtime.h>
#include <hip/hip_bf16.h>

// ---------------- problem constants ----------------
#define CIN   256
#define COUT  256
#define HH    56
#define WW    56
#define BBAT  16
#define KD    (CIN * 9)        // 2304 im2col depth
#define PLANE (HH * WW)        // 3136
#define NPIX  (BBAT * PLANE)   // 50176
#define MEXT  18
#define TABLE 16
#define RBW   2.5f
#define OUTN  (NPIX * COUT)    // 12845056 floats, then 256 int mask

typedef __attribute__((ext_vector_type(16))) _Float16 v16h;
typedef __attribute__((ext_vector_type(8)))  _Float16 v8h;
typedef __attribute__((ext_vector_type(8)))  float    v8f;

// ---------------- ws layout ----------------
//  bytes [0..1023]        : int kidx[256]
//  bytes [1024..1087]     : int cnt[16]
//  bytes [1152..2175]     : float maskf[256]
//  bytes [2176..2179]     : float scale
//  bytes [4096..)         : f16 weights  [256*2304]      (1,179,648 B)
//  bytes [1183744..)      : f16 x        [16*256*3136]   (25,690,112 B)
#define WH_OFF_B   4096u
#define XH_OFF_B   (WH_OFF_B + (unsigned)(COUT * KD * 2))
#define WS_NEED_B  ((size_t)XH_OFF_B + (size_t)NPIX * CIN * 2)

// =============== phase 0: zero vote counters ===============
__global__ void alsh_init_kernel(int* __restrict__ cnt) {
    if (threadIdx.x < TABLE) cnt[threadIdx.x] = 0;
}

// =============== phase 1: hash each filter =================
__global__ __launch_bounds__(256)
void alsh_hash_kernel(const float* __restrict__ w,
                      const float* __restrict__ a,
                      const float* __restrict__ bscal,
                      int* __restrict__ kidx) {
    __shared__ float s1[256];
    __shared__ float s2[256];
    const int k = blockIdx.x, t = threadIdx.x;
    const float* flat = w + k * KD;
    float d = 0.f, q = 0.f;
    for (int j = t; j < KD; j += 256) {
        float v = flat[j];
        d += v * a[j];
        q += v * v;
    }
    s1[t] = d; s2[t] = q;
    __syncthreads();
    for (int s = 128; s > 0; s >>= 1) {
        if (t < s) { s1[t] += s1[t + s]; s2[t] += s2[t + s]; }
        __syncthreads();
    }
    if (t == 0) {
        float suma = 0.f;
        for (int i = 0; i < MEXT; ++i) suma += a[KD + i];
        float h  = s1[0] + (s2[0] / (float)MEXT) * suma + bscal[0];
        float fl = floorf(h / RBW);
        kidx[k]  = (int)fabsf(fmodf(fl, (float)TABLE));
    }
}

// =============== phase 2: per-pixel vote ===================
__global__ __launch_bounds__(256)
void alsh_vote_kernel(const float* __restrict__ x,
                      const float* __restrict__ a,
                      const float* __restrict__ bscal,
                      int* __restrict__ cnt) {
    __shared__ float sa[KD + MEXT];
    const int t = threadIdx.x;
    for (int j = t; j < KD + MEXT; j += 256) sa[j] = a[j];
    __syncthreads();

    const int p   = blockIdx.x * 256 + t;          // 196*256 == NPIX exactly
    const int b   = p / PLANE;
    const int rem = p - b * PLANE;
    const int y   = rem / WW;
    const int xx  = rem - y * WW;
    const float* xb = x + (size_t)b * CIN * PLANE;

    float acc = 0.f;
    for (int r = 0; r < 9; ++r) {
        const int kh = r / 3, kw = r - kh * 3;
        const int iy = y + kh - 1, ix = xx + kw - 1;
        if ((unsigned)iy < (unsigned)HH && (unsigned)ix < (unsigned)WW) {
            acc += 0.5f * (sa[KD + r] + sa[KD + 9 + r]);  // two 0.5 channels
            const int off = iy * WW + ix;
            for (int c = 0; c < CIN; ++c)
                acc += xb[c * PLANE + off] * sa[c * 9 + r];
        }
    }
    const float fl = floorf((acc + bscal[0]) / RBW);
    const int v = (int)fabsf(fmodf(fl, (float)TABLE));
    atomicAdd(&cnt[v], 1);
}

// =============== phase 3: pick bucket, build mask ==========
__global__ __launch_bounds__(256)
void alsh_pick_kernel(const int* __restrict__ kidx,
                      const int* __restrict__ cnt,
                      float* __restrict__ maskf,
                      float* __restrict__ scalep,
                      int* __restrict__ mask_out) {
    __shared__ int sbest;
    __shared__ int sn[256];
    const int t = threadIdx.x;
    if (t == 0) {
        int best = 0, bc = cnt[0];
        for (int v = 1; v < TABLE; ++v)
            if (cnt[v] > bc) { bc = cnt[v]; best = v; }   // first max (argmax)
        sbest = best;
    }
    __syncthreads();
    const int m = (kidx[t] == sbest) ? 1 : 0;
    sn[t] = m;
    __syncthreads();
    for (int s = 128; s > 0; s >>= 1) {
        if (t < s) sn[t] += sn[t + s];
        __syncthreads();
    }
    const int n = sn[0];
    const int empty = (n == 0);
    const float mf = empty ? 1.0f : (float)m;
    maskf[t]    = mf;
    mask_out[t] = (int)mf;
    if (t == 0) scalep[0] = (float)COUT / (float)(empty ? COUT : n);
}

// =============== fp32 -> f16 bulk convert (n % 2048 == 0) ==========
__global__ __launch_bounds__(256)
void alsh_cvt_kernel(const float* __restrict__ src, _Float16* __restrict__ dst) {
    const int i = (blockIdx.x * 256 + threadIdx.x) * 8;
    v8h o;
#pragma unroll
    for (int j = 0; j < 8; ++j) o[j] = (_Float16)src[i + j];
    *(v8h*)(dst + i) = o;
}

// =============== phase 4: masked conv as implicit-GEMM WMMA ==========
// C[256 x 50176] = W[256 x 2304] * im2col(X)[2304 x 50176], f16 in / f32 acc.
// Block: 256 thr = 8 waves (2M x 4N); block tile 64 ko x 128 pixels;
// wave tile 32 x 32 -> 4 x v_wmma_f32_16x16x32_f16 per K-step.
// Double-buffered LDS software pipeline: global loads for step kk+32 are
// issued before the WMMAs of step kk, stores+single barrier afterwards.
#define LDSTR 40   // halves per LDS row (32 K + 8 pad)

template <typename T>
__global__ __launch_bounds__(256)
void alsh_conv_wmma_kernel(const T* __restrict__ xin,
                           const T* __restrict__ wsrc,
                           const float* __restrict__ maskf,
                           const float* __restrict__ scalep,
                           float* __restrict__ out) {
    __shared__ __align__(16) _Float16 As[2][64  * LDSTR];  // [m][k]
    __shared__ __align__(16) _Float16 Bs[2][128 * LDSTR];  // [pixel][k]

    const int t    = threadIdx.x;
    const int p0   = blockIdx.x * 128;    // pixel tile (392*128 == NPIX)
    const int ko0  = blockIdx.y * 64;     // out-channel tile
    const int lane = t & 31;
    const int wv   = t >> 5;
    const int h    = lane >> 4;           // wave half
    const int ln   = lane & 15;
    const int wm2  = wv & 1;              // 2 M groups of 32
    const int wn4  = wv >> 1;             // 4 N groups of 32

    // --- B staging: thread owns 2 k-rows x 8 fixed pixels ---
    const int kr0 = (t >> 4) * 2;         // 0,2,..,30
    const int jb  = (t & 15) * 8;         // pixel group base (0..120)
    int pbase[8], py[8], px[8];
#pragma unroll
    for (int j = 0; j < 8; ++j) {
        const int p   = p0 + jb + j;
        const int b   = p / PLANE;
        const int rem = p - b * PLANE;
        py[j]    = rem / WW;
        px[j]    = rem - py[j] * WW;
        pbase[j] = b * (CIN * PLANE) + rem;
    }
    // --- A staging: thread owns 8 contiguous k of one filter row ---
    const int am = t >> 2;                // 0..63
    const int ak = (t & 3) * 8;

    const v8f vzero = {0.f, 0.f, 0.f, 0.f, 0.f, 0.f, 0.f, 0.f};
    v8f acc[2][2];
    acc[0][0] = vzero; acc[0][1] = vzero;
    acc[1][0] = vzero; acc[1][1] = vzero;

    // ---- prologue: stage K-step 0 into buffer 0 ----
    {
        const T* wr = wsrc + (ko0 + am) * KD + ak;
        v8h tmp;
#pragma unroll
        for (int j = 0; j < 8; ++j) tmp[j] = (_Float16)wr[j];
        *(v8h*)(As[0] + am * LDSTR + ak) = tmp;
#pragma unroll
        for (int kq = 0; kq < 2; ++kq) {
            const int k  = kr0 + kq;
            const int ci = k / 9;
            const int r  = k - ci * 9;
            const int kh = r / 3;
            const int kw = r - kh * 3;
            const int dy = kh - 1, dx = kw - 1;
            const int off = ci * PLANE + dy * WW + dx;
#pragma unroll
            for (int j = 0; j < 8; ++j) {
                const int iy = py[j] + dy, ix = px[j] + dx;
                _Float16 v = (_Float16)0.f;
                if ((unsigned)iy < (unsigned)HH && (unsigned)ix < (unsigned)WW)
                    v = (_Float16)xin[pbase[j] + off];
                Bs[0][(jb + j) * LDSTR + kr0 + kq] = v;
            }
        }
    }
    __syncthreads();

    // ---- pipelined main loop ----
    int buf = 0;
    for (int kk = 0; kk < KD; kk += 32, buf ^= 1) {
        const int  kkn     = kk + 32;
        const bool hasNext = (kkn < KD);

        // (1) issue next tile's global loads into registers
        v8h      aNext;
        _Float16 bNext[16];
        if (hasNext) {
            const T* wr = wsrc + (ko0 + am) * KD + kkn + ak;
            __builtin_prefetch(wr + 32, 0, 1);
#pragma unroll
            for (int j = 0; j < 8; ++j) aNext[j] = (_Float16)wr[j];
#pragma unroll
            for (int kq = 0; kq < 2; ++kq) {
                const int k  = kkn + kr0 + kq;
                const int ci = k / 9;
                const int r  = k - ci * 9;
                const int kh = r / 3;
                const int kw = r - kh * 3;
                const int dy = kh - 1, dx = kw - 1;
                const int off = ci * PLANE + dy * WW + dx;
#pragma unroll
                for (int j = 0; j < 8; ++j) {
                    const int iy = py[j] + dy, ix = px[j] + dx;
                    _Float16 v = (_Float16)0.f;
                    if ((unsigned)iy < (unsigned)HH && (unsigned)ix < (unsigned)WW)
                        v = (_Float16)xin[pbase[j] + off];
                    bNext[kq * 8 + j] = v;
                }
            }
        }

        // (2) fragments + WMMA from current buffer (hides load latency)
        const _Float16* Ab = As[buf];
        const _Float16* Bb = Bs[buf];
        v16h af[2];
#pragma unroll
        for (int amf = 0; amf < 2; ++amf) {
            const _Float16* ap = Ab + (wm2 * 32 + amf * 16 + ln) * LDSTR + 8 * h;
            v8h* dst = (v8h*)&af[amf];
            dst[0] = *(const v8h*)ap;          // elements 0..7  (k = 8h..)
            dst[1] = *(const v8h*)(ap + 16);   // elements 8..15 (k = 16+8h..)
        }
        v16h bf[2];
#pragma unroll
        for (int bnf = 0; bnf < 2; ++bnf) {
            const _Float16* bp = Bb + (wn4 * 32 + bnf * 16 + ln) * LDSTR + 8 * h;
            v8h* dst = (v8h*)&bf[bnf];
            dst[0] = *(const v8h*)bp;
            dst[1] = *(const v8h*)(bp + 16);
        }
#pragma unroll
        for (int amf = 0; amf < 2; ++amf)
#pragma unroll
            for (int bnf = 0; bnf < 2; ++bnf)
                acc[amf][bnf] = __builtin_amdgcn_wmma_f32_16x16x32_f16(
                    false, af[amf], false, bf[bnf], (short)0, acc[amf][bnf],
                    false, false);

        // (3) commit staged registers to the other buffer
        if (hasNext) {
            _Float16* An = As[buf ^ 1];
            _Float16* Bn = Bs[buf ^ 1];
            *(v8h*)(An + am * LDSTR + ak) = aNext;
#pragma unroll
            for (int kq = 0; kq < 2; ++kq)
#pragma unroll
                for (int j = 0; j < 8; ++j)
                    Bn[(jb + j) * LDSTR + kr0 + kq] = bNext[kq * 8 + j];
        }
        __syncthreads();
    }

    // epilogue: D layout -> lane n, VGPR r -> m = r + 8h
    const float sc = scalep[0];
#pragma unroll
    for (int bnf = 0; bnf < 2; ++bnf) {
        const int p   = p0 + wn4 * 32 + bnf * 16 + ln;
        const int b   = p / PLANE;
        const int rem = p - b * PLANE;
#pragma unroll
        for (int amf = 0; amf < 2; ++amf) {
#pragma unroll
            for (int r = 0; r < 8; ++r) {
                const int ko = ko0 + wm2 * 32 + amf * 16 + r + 8 * h;
                const float val = acc[amf][bnf][r] * maskf[ko] * sc;
                out[(size_t)(b * COUT + ko) * PLANE + rem] = val;
            }
        }
    }
}

// ======================= launcher =======================
extern "C" void kernel_launch(void* const* d_in, const int* in_sizes, int n_in,
                              void* d_out, int out_size, void* d_ws, size_t ws_size,
                              hipStream_t stream) {
    (void)in_sizes; (void)n_in; (void)out_size;
    const float* x   = (const float*)d_in[0];   // [16,256,56,56]
    const float* wgt = (const float*)d_in[1];   // [256,256,3,3]
    const float* a   = (const float*)d_in[2];   // [2322]
    const float* bsc = (const float*)d_in[3];   // scalar

    float* outf     = (float*)d_out;
    int*   mask_out = (int*)(outf + OUTN);      // 256 ints after conv output

    char*  wsb    = (char*)d_ws;
    int*   kidx   = (int*)wsb;                  // bytes [0..1023]
    int*   cnt    = (int*)(wsb + 1024);         // bytes [1024..1087]
    float* maskf  = (float*)(wsb + 1152);       // bytes [1152..2175]
    float* scalep = (float*)(wsb + 2176);       // byte  2176

    alsh_init_kernel<<<1, 32, 0, stream>>>(cnt);
    alsh_hash_kernel<<<COUT, 256, 0, stream>>>(wgt, a, bsc, kidx);
    alsh_vote_kernel<<<NPIX / 256, 256, 0, stream>>>(x, a, bsc, cnt);
    alsh_pick_kernel<<<1, 256, 0, stream>>>(kidx, cnt, maskf, scalep, mask_out);

    dim3 grid(NPIX / 128, COUT / 64);           // (392, 4)
    if (ws_size >= WS_NEED_B) {
        // pre-convert w and x to f16 once: halves hot-loop L2 traffic,
        // removes all v_cvt from the WMMA loop
        _Float16* wh = (_Float16*)(wsb + WH_OFF_B);
        _Float16* xh = (_Float16*)(wsb + XH_OFF_B);
        alsh_cvt_kernel<<<(COUT * KD) / 2048, 256, 0, stream>>>(wgt, wh);
        alsh_cvt_kernel<<<(NPIX * CIN) / 2048, 256, 0, stream>>>(x, xh);
        alsh_conv_wmma_kernel<_Float16><<<grid, 256, 0, stream>>>(
            xh, wh, maskf, scalep, outf);
    } else {
        alsh_conv_wmma_kernel<float><<<grid, 256, 0, stream>>>(
            x, wgt, maskf, scalep, outf);
    }
}
